// TreeTransformer_1855425872500
// MI455X (gfx1250) — matrix operand order repaired and checked by reading, not compiled, not run on hardware
//
#include <hip/hip_runtime.h>

// ---------------- problem constants ----------------
#define FBATCH 4096        // B*A
#define NNODE  42
#define NPAD   48
#define NINPUT 32
#define HDIM   128
#define NHEADS 4
#define HEADD  32
#define NLAYER 2
#define FFDIM  256
#define OUTD   128
#define NEDGE  41
#define BFAC   3
#define KOUT   (NNODE*HDIM)   // 5376

typedef _Float16 h16;
typedef __attribute__((ext_vector_type(16))) _Float16 v16h;
typedef __attribute__((ext_vector_type(8)))  _Float16 v8h;
typedef __attribute__((ext_vector_type(8)))  float    v8f;
typedef __attribute__((ext_vector_type(4)))  float    v4f;

// ---------------- WMMA helpers ----------------
__device__ __forceinline__ v8f wmma32(v16h a, v16h b, v8f c) {
  // D = A(16x32 f16) x B(32x16 f16) + C(16x16 f32)
  return __builtin_amdgcn_wmma_f32_16x16x32_f16(false, a, false, b, (short)0, c,
                                                false, false);
}

__device__ __forceinline__ v16h pack16(v8h lo, v8h hi) {
  v16h r;
#pragma unroll
  for (int i = 0; i < 8; ++i) { r[i] = lo[i]; r[i + 8] = hi[i]; }
  return r;
}

// A-matrix 16x32 f16 fragment: lane<16 -> row=lane, K=k0..k0+15 contiguous
// lane>=16 -> row=lane-16, K=k0+16..k0+31
__device__ __forceinline__ v16h frag_a_lds(const h16* base, int stride, int row0, int k0) {
  const int l = threadIdx.x & 31;
  const h16* p = base + (row0 + (l & 15)) * stride + k0 + ((l >> 4) << 4);
  v8h lo = *(const v8h*)(p);
  v8h hi = *(const v8h*)(p + 8);
  return pack16(lo, hi);
}

// B-matrix 32x16 f16 fragment: lane<16 -> col=lane, K=k0..k0+15 contiguous;
// lane>=16 -> col=lane-16, K=k0+16..k0+31.  base is [N][K] layout (weight [out][in]).
__device__ __forceinline__ v16h frag_b_lds(const h16* base, int stride, int n0, int k0) {
  const int l = threadIdx.x & 31;
  const h16* p = base + (n0 + (l & 15)) * stride + k0 + ((l >> 4) << 4);
  v8h lo = *(const v8h*)(p);
  v8h hi = *(const v8h*)(p + 8);
  return pack16(lo, hi);
}

// B fragment directly from global f32 weight stored [N][K] (row = out, contiguous in)
__device__ __forceinline__ v16h frag_b_glb(const float* __restrict__ w, int stride,
                                           int n0, int k0) {
  const int l = threadIdx.x & 31;
  const float* p = w + (size_t)(n0 + (l & 15)) * stride + k0 + ((l >> 4) << 4);
  v16h r;
#pragma unroll
  for (int i = 0; i < 16; i += 4) {
    v4f f = *(const v4f*)(p + i);
    r[i] = (h16)f[0]; r[i + 1] = (h16)f[1]; r[i + 2] = (h16)f[2]; r[i + 3] = (h16)f[3];
  }
  return r;
}

// A fragment from global f16 (row-major, stride in elements)
__device__ __forceinline__ v16h frag_a_glb(const h16* __restrict__ base, size_t stride,
                                           size_t row0, int k0) {
  const int l = threadIdx.x & 31;
  const h16* p = base + (row0 + (size_t)(l & 15)) * stride + k0 + ((l >> 4) << 4);
  v8h lo = *(const v8h*)(p);
  v8h hi = *(const v8h*)(p + 8);
  return pack16(lo, hi);
}

// ---------------- LDS layout (bytes) for encoder kernel ----------------
// xf padded to 48 rows so GEMM epilogues store unconditionally (no exec-mask branches)
#define OFF_XF    0u        // float [48][128]           24576
#define OFF_XH    24576u    // h16   [48][128]           12288
#define OFF_QH    36864u    // h16   [48][128]
#define OFF_KH    49152u    // h16   [48][128]   (attnH/h1H alias 36864..61440)
#define OFF_VT    61440u    // h16   [4][32][64]         16384
#define OFF_S     77824u    // float [4][48][48]         36864 (oH / staging alias)
#define OFF_DEPTH 114688u   // int [48]
#define OFF_EDGE  114880u   // int [123]
#define SMEM_A    115712u

__device__ __forceinline__ void ln_rows(float* xf, h16* xh,
                                        const float* __restrict__ g,
                                        const float* __restrict__ b) {
  const int tid = threadIdx.x;
  if (tid < NNODE) {
    float* row = xf + tid * HDIM;
    float mu = 0.f;
    for (int i = 0; i < HDIM; ++i) mu += row[i];
    mu *= (1.f / HDIM);
    float var = 0.f;
    for (int i = 0; i < HDIM; ++i) { float d = row[i] - mu; var += d * d; }
    var *= (1.f / HDIM);
    float inv = rsqrtf(var + 1e-5f);
    for (int i = 0; i < HDIM; ++i) {
      float v = (row[i] - mu) * inv * g[i] + b[i];
      row[i] = v;
      xh[tid * HDIM + i] = (h16)v;
    }
  }
}

// ==================== Kernel A: per-forest fused encoder ====================
extern "C" __global__ __launch_bounds__(256)
void tree_encoder_kernel(const float* __restrict__ forest,
                         const int*   __restrict__ adjacency,
                         const int*   __restrict__ node_order,
                         const float* __restrict__ W_in,
                         const float* __restrict__ b_in,
                         const float* __restrict__ qkv_w,
                         const float* __restrict__ qkv_b,
                         const float* __restrict__ attn_out_w,
                         const float* __restrict__ attn_out_b,
                         const float* __restrict__ ff1_w,
                         const float* __restrict__ ff1_b,
                         const float* __restrict__ ff2_w,
                         const float* __restrict__ ff2_b,
                         const float* __restrict__ ln1_g,
                         const float* __restrict__ ln1_b,
                         const float* __restrict__ ln2_g,
                         const float* __restrict__ ln2_b,
                         h16* __restrict__ xws) {
  extern __shared__ char smem[];
  float* xf    = (float*)(smem + OFF_XF);   // [48][128] (rows 42..47 = scratch)
  h16*   xh    = (h16*)  (smem + OFF_XH);
  h16*   qH    = (h16*)  (smem + OFF_QH);
  h16*   kH    = (h16*)  (smem + OFF_KH);
  h16*   attnH = (h16*)  (smem + OFF_QH);   // [4][48][64] alias qH+kH
  h16*   h1H   = (h16*)  (smem + OFF_QH);   // [48][256]   alias
  h16*   vT    = (h16*)  (smem + OFF_VT);   // [4][32][64]
  float* S     = (float*)(smem + OFF_S);    // [4][48][48]
  h16*   oH    = (h16*)  (smem + OFF_S);    // [48][128] alias S
  h16*   fH    = (h16*)  (smem + OFF_S);    // [48][32] (step0 staging, alias S)
  h16*   wiH   = (h16*)  (smem + OFF_S + 3072u); // [128][32] (step0 staging)
  int*   depthL = (int*)(smem + OFF_DEPTH);
  int*   edgeL  = (int*)(smem + OFF_EDGE);

  const int tid  = threadIdx.x;
  const int wave = tid >> 5;
  const int lane = tid & 31;
  const int nl   = lane & 15;
  const int hi2  = lane >> 4;
  const int fb   = blockIdx.x;

  // ---- step 0a: zero x, stage forest (f16, padded) + W_in (f16) + edges + depth ----
  for (int i = tid; i < NPAD * HDIM; i += 256) xf[i] = 0.f;
  for (int i = tid; i < NPAD * NINPUT; i += 256) {
    int row = i >> 5, k = i & 31;
    fH[i] = (row < NNODE) ? (h16)forest[((size_t)fb * NNODE + row) * NINPUT + k] : (h16)0.f;
  }
  for (int i = tid; i < HDIM * NINPUT; i += 256) wiH[i] = (h16)W_in[i];
  if (tid < NEDGE * 3) edgeL[tid] = adjacency[(size_t)fb * NEDGE * 3 + tid];
  if (tid == 0) {
    int maxo = -2147483647;
    for (int i = 0; i < NNODE; ++i) {
      int v = node_order[(size_t)fb * NNODE + i];
      maxo = v > maxo ? v : maxo;
    }
    for (int i = 0; i < NNODE; ++i)
      depthL[i] = maxo - node_order[(size_t)fb * NNODE + i];
  }
  __syncthreads();

  // ---- step 0b: tree positional encoding (depth-ordered, matches reference) ----
  for (int d = 0; d < NNODE; ++d) {
    if (tid < HDIM) {
      for (int e = 0; e < NEDGE; ++e) {
        int p = edgeL[e * 3 + 0];
        int c = edgeL[e * 3 + 1];
        int s = edgeL[e * 3 + 2];
        if (p >= 0 && c >= 0) {
          int pl = p - fb * NNODE, cl = c - fb * NNODE;
          if (pl >= 0 && pl < NNODE && cl >= 0 && cl < NNODE && depthL[pl] == d) {
            int si = s + 1; si = si < 0 ? 0 : (si > BFAC - 1 ? BFAC - 1 : si);
            int hot = d * BFAC + si;
            float add = (tid == hot) ? 1.f : 0.f;
            xf[cl * HDIM + tid] = xf[pl * HDIM + tid] + add;
          }
        }
      }
    }
    __syncthreads();
  }

  // ---- step 0c: x += forest @ W_in^T + b_in  (M=48, N=128, K=32; 24 tiles) ----
  for (int t = wave; t < 24; t += 8) {
    int mt = t % 3, nt = t / 3;
    v8f acc;
#pragma unroll
    for (int i = 0; i < 8; ++i) acc[i] = 0.f;
    v16h a = frag_a_lds(fH, NINPUT, mt * 16, 0);
    v16h b = frag_b_lds(wiH, NINPUT, nt * 16, 0);
    acc = wmma32(a, b, acc);
    int col = nt * 16 + nl;
    float bias = b_in[col];
    int mb = mt * 16 + hi2 * 8;
#pragma unroll
    for (int r = 0; r < 8; ++r) xf[(mb + r) * HDIM + col] += acc[r] + bias;
  }
  __syncthreads();

  // xh = f16(x), zero-padded rows 42..47
  for (int i = tid; i < NPAD * HDIM; i += 256) {
    int row = i >> 7;
    xh[i] = (row < NNODE) ? (h16)xf[i] : (h16)0.f;
  }
  __syncthreads();

  // ==================== encoder layers ====================
  for (int l = 0; l < NLAYER; ++l) {
    const float* wq = qkv_w + (size_t)l * 3 * HDIM * HDIM;
    const float* bq = qkv_b + l * 3 * HDIM;
    const float* wo = attn_out_w + (size_t)l * HDIM * HDIM;
    const float* bo = attn_out_b + l * HDIM;
    const float* w1 = ff1_w + (size_t)l * FFDIM * HDIM;
    const float* b1 = ff1_b + l * FFDIM;
    const float* w2 = ff2_w + (size_t)l * HDIM * FFDIM;
    const float* b2 = ff2_b + l * HDIM;

    // zero vT (keys 48..63 must stay 0)
    for (int i = tid; i < NHEADS * HEADD * 64; i += 256) vT[i] = (h16)0.f;
    __syncthreads();

    // ---- QKV: M=48, K=128 (4 kt), N=384 (24 nt) ----
    for (int t = wave; t < 72; t += 8) {
      int mt = t % 3, nt = t / 3;
      v8f acc;
#pragma unroll
      for (int i = 0; i < 8; ++i) acc[i] = 0.f;
#pragma unroll
      for (int kt = 0; kt < 4; ++kt) {
        v16h a = frag_a_lds(xh, HDIM, mt * 16, kt * 32);
        v16h b = frag_b_glb(wq, HDIM, nt * 16, kt * 32);
        acc = wmma32(a, b, acc);
      }
      int col = nt * 16 + nl;
      float bias = bq[col];
      int mb = mt * 16 + hi2 * 8;
      // routing is uniform per tile (tiles never straddle the Q/K/V boundary)
      if (nt < 8) {
#pragma unroll
        for (int r = 0; r < 8; ++r)
          qH[(mb + r) * HDIM + col] = (h16)((acc[r] + bias) * 0.17677669529663688f);
      } else if (nt < 16) {
        int ck = col - HDIM;
#pragma unroll
        for (int r = 0; r < 8; ++r)
          kH[(mb + r) * HDIM + ck] = (h16)(acc[r] + bias);
      } else {
        int c2 = col - 2 * HDIM;
        int hh = c2 >> 5, dd = c2 & 31;
#pragma unroll
        for (int r = 0; r < 8; ++r)
          vT[(hh * HEADD + dd) * 64 + (mb + r)] = (h16)(acc[r] + bias);  // transposed V
      }
    }
    __syncthreads();

    // ---- scores: per head q(48x32) @ k^T(32x48) -> S ----
    for (int t = wave; t < 36; t += 8) {
      int hh = t / 9, rem = t % 9, mt = rem / 3, nt = rem % 3;
      v8f acc;
#pragma unroll
      for (int i = 0; i < 8; ++i) acc[i] = 0.f;
      v16h a = frag_a_lds(qH, HDIM, mt * 16, hh * HEADD);
      v16h b = frag_b_lds(kH, HDIM, nt * 16, hh * HEADD);
      acc = wmma32(a, b, acc);
      int key = nt * 16 + nl;
      int mb = mt * 16 + hi2 * 8;
#pragma unroll
      for (int r = 0; r < 8; ++r) S[(hh * NPAD + mb + r) * NPAD + key] = acc[r];
    }
    __syncthreads();

    // ---- softmax over 42 keys -> attnH (f16, K padded to 64 with zeros) ----
    if (tid < NHEADS * NPAD) {
      int hh = tid / NPAD, m = tid % NPAD;
      float* srow = S + (hh * NPAD + m) * NPAD;
      h16* arow = attnH + (hh * NPAD + m) * 64;
      if (m >= NNODE) {
        for (int k = 0; k < 64; ++k) arow[k] = (h16)0.f;
      } else {
        float mx = -1e30f;
        for (int k = 0; k < NNODE; ++k) mx = fmaxf(mx, srow[k]);
        float sum = 0.f;
        for (int k = 0; k < NNODE; ++k) { float e = __expf(srow[k] - mx); srow[k] = e; sum += e; }
        float inv = 1.f / sum;
        for (int k = 0; k < NNODE; ++k) arow[k] = (h16)(srow[k] * inv);
        for (int k = NNODE; k < 64; ++k) arow[k] = (h16)0.f;
      }
    }
    __syncthreads();

    // ---- attn @ V: per head (48x64)@(64x32) -> oH[48][128] ----
    for (int t = wave; t < 24; t += 8) {
      int hh = t / 6, rem = t % 6, mt = rem / 2, dt = rem % 2;
      v8f acc;
#pragma unroll
      for (int i = 0; i < 8; ++i) acc[i] = 0.f;
#pragma unroll
      for (int kt = 0; kt < 2; ++kt) {
        v16h a = frag_a_lds(attnH + hh * NPAD * 64, 64, mt * 16, kt * 32);
        v16h b = frag_b_lds(vT + hh * HEADD * 64, 64, dt * 16, kt * 32);
        acc = wmma32(a, b, acc);
      }
      int col = hh * HEADD + dt * 16 + nl;
      int mb = mt * 16 + hi2 * 8;
#pragma unroll
      for (int r = 0; r < 8; ++r) oH[(mb + r) * HDIM + col] = (h16)acc[r];
    }
    __syncthreads();

    // ---- out-proj: o @ attn_out_w^T + residual ----
    for (int t = wave; t < 24; t += 8) {
      int mt = t % 3, nt = t / 3;
      v8f acc;
#pragma unroll
      for (int i = 0; i < 8; ++i) acc[i] = 0.f;
#pragma unroll
      for (int kt = 0; kt < 4; ++kt) {
        v16h a = frag_a_lds(oH, HDIM, mt * 16, kt * 32);
        v16h b = frag_b_glb(wo, HDIM, nt * 16, kt * 32);
        acc = wmma32(a, b, acc);
      }
      int col = nt * 16 + nl;
      float bias = bo[col];
      int mb = mt * 16 + hi2 * 8;
#pragma unroll
      for (int r = 0; r < 8; ++r) xf[(mb + r) * HDIM + col] += acc[r] + bias;
    }
    __syncthreads();

    ln_rows(xf, xh, ln1_g + l * HDIM, ln1_b + l * HDIM);
    __syncthreads();

    // ---- FF1: relu(x @ ff1_w^T + b1) -> h1H[48][256] ----
    for (int t = wave; t < 48; t += 8) {
      int mt = t % 3, nt = t / 3;
      v8f acc;
#pragma unroll
      for (int i = 0; i < 8; ++i) acc[i] = 0.f;
#pragma unroll
      for (int kt = 0; kt < 4; ++kt) {
        v16h a = frag_a_lds(xh, HDIM, mt * 16, kt * 32);
        v16h b = frag_b_glb(w1, HDIM, nt * 16, kt * 32);
        acc = wmma32(a, b, acc);
      }
      int col = nt * 16 + nl;
      float bias = b1[col];
      int mb = mt * 16 + hi2 * 8;
#pragma unroll
      for (int r = 0; r < 8; ++r)
        h1H[(mb + r) * FFDIM + col] = (h16)fmaxf(acc[r] + bias, 0.f);
    }
    __syncthreads();

    // ---- FF2: h1 @ ff2_w^T + residual ----
    for (int t = wave; t < 24; t += 8) {
      int mt = t % 3, nt = t / 3;
      v8f acc;
#pragma unroll
      for (int i = 0; i < 8; ++i) acc[i] = 0.f;
#pragma unroll
      for (int kt = 0; kt < 8; ++kt) {
        v16h a = frag_a_lds(h1H, FFDIM, mt * 16, kt * 32);
        v16h b = frag_b_glb(w2, FFDIM, nt * 16, kt * 32);
        acc = wmma32(a, b, acc);
      }
      int col = nt * 16 + nl;
      float bias = b2[col];
      int mb = mt * 16 + hi2 * 8;
#pragma unroll
      for (int r = 0; r < 8; ++r) xf[(mb + r) * HDIM + col] += acc[r] + bias;
    }
    __syncthreads();

    ln_rows(xf, xh, ln2_g + l * HDIM, ln2_b + l * HDIM);
    __syncthreads();
  }

  // ---- store encoder output as f16 for the output GEMM ----
  for (int i = tid; i < NNODE * HDIM; i += 256)
    xws[(size_t)fb * KOUT + i] = (h16)xf[i];
}

// ==================== Kernel B: out = LN(x @ W_out^T + b_out) ====================
// M=4096 (64 rows/block), N=128 (8 n-tiles), K=5376 (168 k-tiles)
#define BROWS 64
#define SMEM_B (BROWS * 132 * 4)

extern "C" __global__ __launch_bounds__(256)
void out_gemm_kernel(const h16*  __restrict__ xws,
                     const float* __restrict__ W_out,
                     const float* __restrict__ b_out,
                     const float* __restrict__ lnf_g,
                     const float* __restrict__ lnf_b,
                     float* __restrict__ out) {
  extern __shared__ char smem[];
  float* outT = (float*)smem;  // [64][132]
  const int tid  = threadIdx.x;
  const int wave = tid >> 5;
  const int lane = tid & 31;
  const int nl   = lane & 15;
  const int hi2  = lane >> 4;
  const size_t row0 = (size_t)blockIdx.x * BROWS;
  const int mt = wave & 3;         // 4 m-tiles of 16
  const int ng = (wave >> 2) * 4;  // n-tile group: waves 0-3 -> nt 0..3, waves 4-7 -> nt 4..7

  v8f acc[4];
#pragma unroll
  for (int j = 0; j < 4; ++j)
#pragma unroll
    for (int i = 0; i < 8; ++i) acc[j][i] = 0.f;

  for (int kt = 0; kt < KOUT / 32; ++kt) {
    v16h a = frag_a_glb(xws, KOUT, row0 + mt * 16, kt * 32);
#pragma unroll
    for (int j = 0; j < 4; ++j) {
      v16h b = frag_b_glb(W_out, KOUT, (ng + j) * 16, kt * 32);
      acc[j] = wmma32(a, b, acc[j]);
    }
  }

#pragma unroll
  for (int j = 0; j < 4; ++j) {
    int col = (ng + j) * 16 + nl;
    float bias = b_out[col];
    int mb = mt * 16 + hi2 * 8;
#pragma unroll
    for (int r = 0; r < 8; ++r) outT[(mb + r) * 132 + col] = acc[j][r] + bias;
  }
  __syncthreads();

  // fused final LayerNorm over OUT=128
  if (tid < BROWS) {
    float* row = outT + tid * 132;
    float mu = 0.f;
    for (int n = 0; n < OUTD; ++n) mu += row[n];
    mu *= (1.f / OUTD);
    float var = 0.f;
    for (int n = 0; n < OUTD; ++n) { float d = row[n] - mu; var += d * d; }
    var *= (1.f / OUTD);
    float inv = rsqrtf(var + 1e-5f);
    for (int n = 0; n < OUTD; ++n)
      out[(row0 + tid) * OUTD + n] = (row[n] - mu) * inv * lnf_g[n] + lnf_b[n];
  }
}

// ==================== host launch ====================
extern "C" void kernel_launch(void* const* d_in, const int* in_sizes, int n_in,
                              void* d_out, int out_size, void* d_ws, size_t ws_size,
                              hipStream_t stream) {
  const float* forest     = (const float*)d_in[0];
  const int*   adjacency  = (const int*)  d_in[1];
  const int*   node_order = (const int*)  d_in[2];
  const float* W_in       = (const float*)d_in[3];
  const float* b_in       = (const float*)d_in[4];
  const float* qkv_w      = (const float*)d_in[5];
  const float* qkv_b      = (const float*)d_in[6];
  const float* attn_out_w = (const float*)d_in[7];
  const float* attn_out_b = (const float*)d_in[8];
  const float* ff1_w      = (const float*)d_in[9];
  const float* ff1_b      = (const float*)d_in[10];
  const float* ff2_w      = (const float*)d_in[11];
  const float* ff2_b      = (const float*)d_in[12];
  const float* ln1_g      = (const float*)d_in[13];
  const float* ln1_b      = (const float*)d_in[14];
  const float* ln2_g      = (const float*)d_in[15];
  const float* ln2_b      = (const float*)d_in[16];
  const float* W_out      = (const float*)d_in[17];
  const float* b_out      = (const float*)d_in[18];
  const float* lnf_g      = (const float*)d_in[19];
  const float* lnf_b      = (const float*)d_in[20];

  h16* xws = (h16*)d_ws;  // FB x 5376 f16 = 44 MB

  tree_encoder_kernel<<<FBATCH, 256, SMEM_A, stream>>>(
      forest, adjacency, node_order, W_in, b_in, qkv_w, qkv_b, attn_out_w,
      attn_out_b, ff1_w, ff1_b, ff2_w, ff2_b, ln1_g, ln1_b, ln2_g, ln2_b, xws);

  out_gemm_kernel<<<FBATCH / BROWS, 256, SMEM_B, stream>>>(
      xws, W_out, b_out, lnf_g, lnf_b, (float*)d_out);
}
